// MultiHeadAttention_16295105921213
// MI455X (gfx1250) — matrix operand order repaired and checked by reading, not compiled, not run on hardware
//
#include <hip/hip_runtime.h>

// ---------------------------------------------------------------------------
// Problem constants (from reference): B=8, T=4096, D=1024, H=16, DH=64
// ---------------------------------------------------------------------------
#define B_    8
#define T_    4096
#define D_    1024
#define H_    16
#define DH_   64
#define M_    (B_ * T_)          // 32768 tokens
#define NC_   16                 // scan chunks
#define CHUNK_ (T_ / NC_)        // 256 steps per chunk

typedef __attribute__((ext_vector_type(16))) _Float16 v16h;
typedef __attribute__((ext_vector_type(8)))  _Float16 v8h;
typedef __attribute__((ext_vector_type(8)))  float    v8f;

// Async global -> LDS copy of 16 bytes (CDNA5 GLOBAL_LOAD_ASYNC_TO_LDS_B128,
// GV addressing: 64-bit VGPR address). Tracked with ASYNCcnt.
__device__ __forceinline__ void async_copy_b128(unsigned lds_byte_addr,
                                                const void* gptr) {
  asm volatile("global_load_async_to_lds_b128 %0, %1, off"
               :: "v"(lds_byte_addr), "v"((unsigned long long)gptr)
               : "memory");
}

// ---------------------------------------------------------------------------
// fp32 -> f16 conversion (grid-stride)
// ---------------------------------------------------------------------------
__global__ void cvt_f32_to_f16(const float* __restrict__ src,
                               _Float16* __restrict__ dst, int n) {
  int i = blockIdx.x * blockDim.x + threadIdx.x;
  int stride = gridDim.x * blockDim.x;
  for (; i < n; i += stride) dst[i] = (_Float16)src[i];
}

// ---------------------------------------------------------------------------
// Fused triple GEMM + activation epilogue.
//   out_v[m,n]     = tanh( sum_k x[m,k] Wr[n,k] + br[n] )
//   out_alpha[m,n] = sigmoid(.. Wf ..) * (1 - sigmoid(.. Wa ..))
// A: [M,K] f16 row-major.  W: [3,N,K] f16 (reach, forget, abs).
// Block tile: 128(M) x 64(N), BK = 32, 8 waves (4 M x 2 N), wave tile 32x32.
// Staging: double-buffered LDS filled by async global->LDS DMA, one K-slab
// prefetched ahead (5 async ops per wave per slab -> s_wait_asynccnt 0x5).
// ---------------------------------------------------------------------------
__launch_bounds__(256)
__global__ void gemm_fused(const _Float16* __restrict__ A,
                           const _Float16* __restrict__ W,
                           const float* __restrict__ br,
                           const float* __restrict__ bf,
                           const float* __restrict__ ba,
                           float* __restrict__ vbuf,
                           float* __restrict__ abuf) {
  constexpr int BM = 128, BN = 64, BK = 32, SA = 40, SB = 40, K = D_, N = D_;
  constexpr int NKSTEP = K / BK;         // 32
  __shared__ _Float16 As[2][BM * SA];    // 2 x 10 KB
  __shared__ _Float16 Bs[2][3 * BN * SB];// 2 x 15 KB

  const int tid   = threadIdx.x;
  const int lane  = tid & 31;
  const int wave  = tid >> 5;
  const int waveM = wave & 3;            // 0..3
  const int waveN = wave >> 2;           // 0..1
  const int blockM = blockIdx.x * BM;
  const int blockN = blockIdx.y * BN;
  const int lhalf = lane >> 4;           // 0 or 1
  const int l16   = lane & 15;

  // LDS byte base addresses (low 32 bits of generic pointer == LDS offset)
  const unsigned asBase0 = (unsigned)(unsigned long long)(void*)&As[0][0];
  const unsigned asBase1 = (unsigned)(unsigned long long)(void*)&As[1][0];
  const unsigned bsBase0 = (unsigned)(unsigned long long)(void*)&Bs[0][0];
  const unsigned bsBase1 = (unsigned)(unsigned long long)(void*)&Bs[1][0];

  // Stage one K-slab (kb = k offset) into buffer `buf` via async DMA.
  auto stage = [&](int kb, int buf) {
    unsigned asB = buf ? asBase1 : asBase0;
    unsigned bsB = buf ? bsBase1 : bsBase0;
    // A tile: 128x32 halves = 512 chunks of 8 halves; 2 per thread
#pragma unroll
    for (int i = 0; i < 2; ++i) {
      int c = tid + i * 256;
      int row = c >> 2;
      int within = (c & 3) * 8;
      async_copy_b128(asB + (unsigned)(row * SA + within) * 2,
                      &A[(size_t)(blockM + row) * K + kb + within]);
    }
    // B tiles: 3 x 64 x 32 halves = 768 chunks; 3 per thread
#pragma unroll
    for (int i = 0; i < 3; ++i) {
      int c = tid + i * 256;
      int row = c >> 2;                  // 0..191
      int within = (c & 3) * 8;
      int g = row >> 6;
      int r = row & 63;
      async_copy_b128(bsB + (unsigned)((g * BN + r) * SB + within) * 2,
                      &W[(size_t)g * N * K + (size_t)(blockN + r) * K + kb + within]);
    }
  };

  v8f acc[3][2][2] = {};

  stage(0, 0);                           // prologue: slab 0 -> buf 0

  for (int ks = 0; ks < NKSTEP; ++ks) {
    const int cur = ks & 1;
    // Prefetch next slab into the other buffer (wrapped dummy on last iter so
    // the in-flight count stays constant). Safe: that buffer's readers all
    // finished before the trailing barrier of the previous iteration.
    const int nxt = (ks + 1 < NKSTEP) ? (ks + 1) : 0;
    stage(nxt * BK, cur ^ 1);
    // Wait for the CURRENT slab only (async loads complete in order): 5 newer
    // ops (just issued) may remain outstanding -> overlap DMA with WMMA.
    asm volatile("s_wait_asynccnt 0x5" ::: "memory");
    __syncthreads();

    const _Float16* as = &As[cur][0];
    const _Float16* bs = &Bs[cur][0];

    // ---- A fragments (16x32 f16: lanes 0-15 K0-7/K16-23, 16-31 K8-15/K24-31)
    v16h afrag[2];
#pragma unroll
    for (int mi = 0; mi < 2; ++mi) {
      int m = waveM * 32 + mi * 16 + l16;
      int koff = lhalf * 8;
      v16h a;
      ((v8h*)&a)[0] = *(const v8h*)&as[m * SA + koff];
      ((v8h*)&a)[1] = *(const v8h*)&as[m * SA + koff + 16];
      afrag[mi] = a;
    }

    // ---- B fragments + WMMA (B 32x16: lanes 0-15 K0-7/K16-23, 16-31 K8-15/K24-31)
#pragma unroll
    for (int g = 0; g < 3; ++g) {
#pragma unroll
      for (int ni = 0; ni < 2; ++ni) {
        int r = waveN * 32 + ni * 16 + l16;
        int khalf = lhalf * 8;
        v16h b;
        ((v8h*)&b)[0] = *(const v8h*)&bs[(g * BN + r) * SB + khalf];
        ((v8h*)&b)[1] = *(const v8h*)&bs[(g * BN + r) * SB + khalf + 16];
#pragma unroll
        for (int mi = 0; mi < 2; ++mi) {
          acc[g][mi][ni] = __builtin_amdgcn_wmma_f32_16x16x32_f16(
              false, afrag[mi], false, b, (short)0, acc[g][mi][ni], false, false);
        }
      }
    }
    __syncthreads();   // readers of buf[cur] done -> next iter may overwrite
  }

  // ---- epilogue: bias + tanh / sigmoid, write v and alpha (f32) ----
#pragma unroll
  for (int ni = 0; ni < 2; ++ni) {
    int n = blockN + waveN * 32 + ni * 16 + l16;
    float brn = br[n], bfn = bf[n], ban = ba[n];
#pragma unroll
    for (int mi = 0; mi < 2; ++mi) {
      int mbase = blockM + waveM * 32 + mi * 16 + lhalf * 8;
#pragma unroll
      for (int r = 0; r < 8; ++r) {
        int m = mbase + r;
        float xr = acc[0][mi][ni][r] + brn;
        float xf = acc[1][mi][ni][r] + bfn;
        float xa = acc[2][mi][ni][r] + ban;
        float vv = 1.f - 2.f / (__expf(2.f * xr) + 1.f);     // tanh
        float sf = 1.f / (1.f + __expf(-xf));                // sigmoid(m)
        float sa = 1.f / (1.f + __expf(-xa));                // sigmoid(s)
        size_t off = (size_t)m * D_ + n;
        vbuf[off] = vv;
        abuf[off] = sf * (1.f - sa);
      }
    }
  }
}

// ---------------------------------------------------------------------------
// Pass 1: per (b,h,chunk) wave computes A = prod(a), S = scan-from-zero tail.
// One wave32 per block; each lane owns 2 contiguous channels (DH=64).
// ---------------------------------------------------------------------------
__global__ void scan_pass1(const float* __restrict__ abuf,
                           const float* __restrict__ vbuf,
                           float* __restrict__ cA, float* __restrict__ cB) {
  int blk = blockIdx.x;
  int c  = blk % NC_;
  int bh = blk / NC_;
  int h  = bh % H_;
  int b  = bh / H_;
  int lane = threadIdx.x;
  int ch = lane * 2;
  size_t base = ((size_t)b * T_ + (size_t)c * CHUNK_) * D_ + h * DH_ + ch;
  float2 A = make_float2(1.f, 1.f);
  float2 S = make_float2(0.f, 0.f);
#pragma unroll 4
  for (int t = 0; t < CHUNK_; ++t) {
    float2 a = *(const float2*)&abuf[base + (size_t)t * D_];
    float2 v = *(const float2*)&vbuf[base + (size_t)t * D_];
    S.x = S.x * a.x + (1.f - a.x) * v.x;
    S.y = S.y * a.y + (1.f - a.y) * v.y;
    A.x *= a.x; A.y *= a.y;
  }
  size_t co = ((size_t)bh * NC_ + c) * DH_ + ch;
  *(float2*)&cA[co] = A;
  *(float2*)&cB[co] = S;
}

// ---------------------------------------------------------------------------
// Pass 2: combine 16 chunk summaries sequentially -> initial state per chunk.
// ---------------------------------------------------------------------------
__global__ void scan_pass2(const float* __restrict__ cA,
                           const float* __restrict__ cB,
                           float* __restrict__ sInit) {
  int bh = blockIdx.x;
  int lane = threadIdx.x;
  int ch = lane * 2;
  float2 s = make_float2(0.f, 0.f);
  for (int c = 0; c < NC_; ++c) {
    size_t co = ((size_t)bh * NC_ + c) * DH_ + ch;
    *(float2*)&sInit[co] = s;
    float2 A  = *(const float2*)&cA[co];
    float2 Bv = *(const float2*)&cB[co];
    s.x = A.x * s.x + Bv.x;
    s.y = A.y * s.y + Bv.y;
  }
}

// ---------------------------------------------------------------------------
// Pass 3: replay each chunk from its true initial state; RMS norm per step via
// wave32 butterfly reduction (64 channels = 2 per lane, no LDS barrier).
// ---------------------------------------------------------------------------
__global__ void scan_pass3(const float* __restrict__ abuf,
                           const float* __restrict__ vbuf,
                           const float* __restrict__ sInit,
                           const float* __restrict__ scale,
                           const float* __restrict__ resg,
                           float* __restrict__ out) {
  int blk = blockIdx.x;
  int c  = blk % NC_;
  int bh = blk / NC_;
  int h  = bh % H_;
  int b  = bh / H_;
  int lane = threadIdx.x;
  int ch = lane * 2;
  size_t base = ((size_t)b * T_ + (size_t)c * CHUNK_) * D_ + h * DH_ + ch;
  float2 s  = *(const float2*)&sInit[((size_t)bh * NC_ + c) * DH_ + ch];
  float2 sc = *(const float2*)&scale[h * DH_ + ch];
  float2 rg = *(const float2*)&resg[h * DH_ + ch];
#pragma unroll 2
  for (int t = 0; t < CHUNK_; ++t) {
    float2 a = *(const float2*)&abuf[base + (size_t)t * D_];
    float2 v = *(const float2*)&vbuf[base + (size_t)t * D_];
    s.x = s.x * a.x + (1.f - a.x) * v.x;
    s.y = s.y * a.y + (1.f - a.y) * v.y;
    float sq = s.x * s.x + s.y * s.y;
#pragma unroll
    for (int off = 16; off > 0; off >>= 1) sq += __shfl_xor(sq, off, 32);
    float rr = rsqrtf(sq * (1.f / DH_) + 1e-12f);
    float2 o;
    o.x = s.x * rr * sc.x + v.x * rg.x;
    o.y = s.y * rr * sc.y + v.y * rg.y;
    *(float2*)&out[base + (size_t)t * D_] = o;
  }
}

// ---------------------------------------------------------------------------
// Launch
// ---------------------------------------------------------------------------
extern "C" void kernel_launch(void* const* d_in, const int* in_sizes, int n_in,
                              void* d_out, int out_size, void* d_ws, size_t ws_size,
                              hipStream_t stream) {
  const float* x     = (const float*)d_in[0];
  const float* Wr    = (const float*)d_in[1];
  const float* br    = (const float*)d_in[2];
  const float* Wf    = (const float*)d_in[3];
  const float* bf    = (const float*)d_in[4];
  const float* Wa    = (const float*)d_in[5];
  const float* ba    = (const float*)d_in[6];
  const float* scale = (const float*)d_in[7];
  const float* rg    = (const float*)d_in[8];
  float* out = (float*)d_out;

  char* ws = (char*)d_ws;
  size_t offA  = 0;                                  // x in f16: 64 MB
  size_t szA   = (size_t)M_ * D_ * 2;
  size_t offW  = offA + szA;                         // W in f16: 6 MB
  size_t szW   = (size_t)3 * D_ * D_ * 2;
  size_t offV  = offW + szW;                         // v buf f32: 128 MB
  size_t szV   = (size_t)M_ * D_ * 4;
  size_t offAl = offV + szV;                         // alpha buf f32: 128 MB
  size_t offcA = offAl + szV;                        // chunk prod: 512 KB
  size_t szc   = (size_t)B_ * H_ * NC_ * DH_ * 4;
  size_t offcB = offcA + szc;
  size_t offsI = offcB + szc;

  _Float16* A16 = (_Float16*)(ws + offA);
  _Float16* W16 = (_Float16*)(ws + offW);
  float* vbuf = (float*)(ws + offV);
  float* abuf = (float*)(ws + offAl);
  float* cA   = (float*)(ws + offcA);
  float* cB   = (float*)(ws + offcB);
  float* sI   = (float*)(ws + offsI);

  cvt_f32_to_f16<<<4096, 256, 0, stream>>>(x, A16, M_ * D_);
  cvt_f32_to_f16<<<1024, 256, 0, stream>>>(Wr, W16 + (size_t)0 * D_ * D_, D_ * D_);
  cvt_f32_to_f16<<<1024, 256, 0, stream>>>(Wf, W16 + (size_t)1 * D_ * D_, D_ * D_);
  cvt_f32_to_f16<<<1024, 256, 0, stream>>>(Wa, W16 + (size_t)2 * D_ * D_, D_ * D_);

  dim3 grid(M_ / 128, D_ / 64);
  gemm_fused<<<grid, 256, 0, stream>>>(A16, W16, br, bf, ba, vbuf, abuf);

  scan_pass1<<<B_ * H_ * NC_, 32, 0, stream>>>(abuf, vbuf, cA, cB);
  scan_pass2<<<B_ * H_, 32, 0, stream>>>(cA, cB, sI);
  scan_pass3<<<B_ * H_ * NC_, 32, 0, stream>>>(abuf, vbuf, sI, scale, rg, out);
}
